// HippoSSKernel_28853590294576
// MI455X (gfx1250) — compile-verified, matrix operand order checked by reading, not run on hardware
//
#include <hip/hip_runtime.h>

typedef _Float16 v16h  __attribute__((ext_vector_type(16)));
typedef _Float16 half8 __attribute__((ext_vector_type(8)));
typedef float    v8f   __attribute__((ext_vector_type(8)));

#define SSD_H 1024
#define SSD_N 64
#define SSD_L 2048

// One workgroup per head h.  128 threads = 4 wave32.
// Phase 1-3 (threads 0..63): build A (16x128 f16) and B^T (128x128 f16) in LDS
//   A[r][2n]   =  Re(Cw * z^r),  A[r][2n+1] = -Im(Cw * z^r)
//   Bt[q][2n]  =  Re(z16^q),     Bt[q][2n+1] =  Im(z16^q)      (z16 = z^16)
// Phase 4: each wave computes two 16x16 output tiles via 4x v_wmma_f32_16x16x32_f16
//   out[h, 16q + r] = 2 * (A . B)[r, q]
__global__ __launch_bounds__(128) void s4_vandermonde_wmma(
    const float* __restrict__ log_dt,
    const float* __restrict__ log_w_real,
    const float* __restrict__ w_imag,
    const float* __restrict__ C_re,
    const float* __restrict__ C_im,
    float* __restrict__ out)
{
  __shared__ __align__(16) _Float16 Ah [16 * 128];    //  4 KB
  __shared__ __align__(16) _Float16 Bth[128 * 128];   // 32 KB (B transposed: [q][n'])

  const int h   = blockIdx.x;
  const int tid = threadIdx.x;

  if (tid < SSD_N) {
    const int n  = tid;
    const float dt = __expf(log_dt[h]);
    const float wr = -__expf(log_w_real[h * SSD_N + n]);
    const float wi = w_imag[h * SSD_N + n];
    const float aR = wr * dt, aI = wi * dt;           // dtA
    float s, c;
    __sincosf(aI, &s, &c);
    const float em = __expf(aR);
    const float zr = em * c, zi = em * s;             // z = exp(dtA), |z| < 1

    // Cw = C * (z - 1) / w   (complex divide via conj(w)/|w|^2)
    const float cr = C_re[h * SSD_N + n], ci = C_im[h * SSD_N + n];
    const float mr = zr - 1.0f, mi = zi;
    const float nr = cr * mr - ci * mi;
    const float ni = cr * mi + ci * mr;
    const float inv = 1.0f / (wr * wr + wi * wi);
    const float cwr = (nr * wr + ni * wi) * inv;
    const float cwi = (ni * wr - nr * wi) * inv;

    // A rows: Cw * z^r, r = 0..15 (iterated complex multiply, f32 regs)
    float pr = cwr, pi = cwi;
    #pragma unroll
    for (int r = 0; r < 16; ++r) {
      Ah[r * 128 + 2 * n]     = (_Float16)pr;
      Ah[r * 128 + 2 * n + 1] = (_Float16)(-pi);
      const float tr = pr * zr - pi * zi;
      pi = pr * zi + pi * zr;
      pr = tr;
    }

    // z16 = z^16 by 4 squarings
    float qr = zr, qi = zi;
    #pragma unroll
    for (int sq = 0; sq < 4; ++sq) {
      const float tr = qr * qr - qi * qi;
      qi = 2.0f * qr * qi;
      qr = tr;
    }

    // B^T rows: z16^q, q = 0..127
    pr = 1.0f; pi = 0.0f;
    for (int q = 0; q < 128; ++q) {
      Bth[q * 128 + 2 * n]     = (_Float16)pr;
      Bth[q * 128 + 2 * n + 1] = (_Float16)pi;
      const float tr = pr * qr - pi * qi;
      pi = pr * qi + pi * qr;
      pr = tr;
    }
  }

  __syncthreads();

  // ---- WMMA phase: all 128 threads active, uniform control flow ----
  const int wave = tid >> 5;
  const int lane = tid & 31;
  const int hi   = lane >> 4;     // lane half selects K+8 (and M/N+8 for C/D)
  const int row  = lane & 15;     // A-fragment M row / B-fragment N column

  union V16 { v16h v; half8 h8[2]; };

  for (int t = 0; t < 2; ++t) {
    const int qt   = wave * 2 + t;     // q-tile 0..7 (8 tiles of 16 q columns)
    const int qcol = qt * 16 + row;    // this lane's q column
    v8f acc = {};
    #pragma unroll
    for (int kc = 0; kc < 4; ++kc) {   // K = 128 in chunks of 32
      // 16-bit A layout: vgpr v<4 holds K = kc*32 + hi*8 + 2v{,+1},
      //                  vgpr v>=4 holds K = kc*32 + 16 + hi*8 + ...
      // => two contiguous 8-half (16B) LDS reads per fragment.
      const int base = kc * 32 + hi * 8;
      V16 a, b;
      a.h8[0] = *(const half8*)(Ah  + row  * 128 + base);
      a.h8[1] = *(const half8*)(Ah  + row  * 128 + base + 16);
      b.h8[0] = *(const half8*)(Bth + qcol * 128 + base);
      b.h8[1] = *(const half8*)(Bth + qcol * 128 + base + 16);
      acc = __builtin_amdgcn_wmma_f32_16x16x32_f16(
          /*neg_a=*/false, a.v, /*neg_b=*/false, b.v,
          /*c_mod=*/(short)0, acc, /*reuse_a=*/false, /*reuse_b=*/false);
    }
    // C/D layout: vgpr i, lane -> M = i + 8*hi (= r), N = lane&15 (= q within tile)
    float* o = out + h * SSD_L;
    #pragma unroll
    for (int i = 0; i < 8; ++i) {
      const int r = i + hi * 8;
      o[qcol * 16 + r] = 2.0f * acc[i];
    }
  }
}

extern "C" void kernel_launch(void* const* d_in, const int* in_sizes, int n_in,
                              void* d_out, int out_size, void* d_ws, size_t ws_size,
                              hipStream_t stream) {
  const float* log_dt     = (const float*)d_in[0];
  const float* log_w_real = (const float*)d_in[1];
  const float* w_imag     = (const float*)d_in[2];
  const float* C_re       = (const float*)d_in[3];
  const float* C_im       = (const float*)d_in[4];
  float* out = (float*)d_out;
  s4_vandermonde_wmma<<<SSD_H, 128, 0, stream>>>(
      log_dt, log_w_real, w_imag, C_re, C_im, out);
}
